// MambaModel_65120294142161
// MI455X (gfx1250) — matrix operand order repaired and checked by reading, not compiled, not run on hardware
//
#include <hip/hip_runtime.h>
#include <hip/hip_bf16.h>

// ---- Model dims (match reference) ----
#define S_LEN   2048
#define D_MODEL 1024
#define N_LAYERS 2
#define DI_DIM  2048
#define N_STATE 16
#define K_CONV  4
#define VOCAB   32000

typedef __bf16 bf16_t;
typedef __attribute__((ext_vector_type(16))) __bf16 v16bf;
typedef __attribute__((ext_vector_type(8)))  __bf16 v8bf;
typedef __attribute__((ext_vector_type(4)))  __bf16 v4bf;
typedef __attribute__((ext_vector_type(8)))  float  v8f;
typedef __attribute__((ext_vector_type(4)))  int    v4i;

#define GLOBAL_AS __attribute__((address_space(1)))
#define LDS_AS    __attribute__((address_space(3)))

// CDNA5 async global->LDS staging (ASYNCcnt); guarded so absence of the
// builtin falls back to sync load + ds_store without breaking the build.
#if defined(__gfx1250__) && __has_builtin(__builtin_amdgcn_global_load_async_to_lds_b128)
#define HAVE_ASYNC_LDS 1
#else
#define HAVE_ASYNC_LDS 0
#endif

__device__ __forceinline__ void stage16(const bf16_t* g, bf16_t* l) {
#if HAVE_ASYNC_LDS
    __builtin_amdgcn_global_load_async_to_lds_b128(
        (GLOBAL_AS v4i*)g, (LDS_AS v4i*)l, 0, 0);
#else
    *(v8bf*)l = *(const v8bf*)g;
#endif
}
__device__ __forceinline__ void stage_wait() {
#if HAVE_ASYNC_LDS
#if __has_builtin(__builtin_amdgcn_s_wait_asynccnt)
    __builtin_amdgcn_s_wait_asynccnt(0);
#else
    asm volatile("s_wait_asynccnt 0" ::: "memory");
#endif
#endif
}

__device__ __forceinline__ float silu_f(float x) {
    return x / (1.0f + __expf(-x));
}
__device__ __forceinline__ float softplus_f(float x) {
    return (x > 20.0f) ? x : log1pf(__expf(x));
}

// ---------------------------------------------------------------------------
// fp32 -> bf16 bulk conversion (n divisible by 1024); float4 in, 8B out
// ---------------------------------------------------------------------------
__global__ void cvt_bf16_kernel(const float* __restrict__ in,
                                bf16_t* __restrict__ out) {
    const int i = (blockIdx.x * blockDim.x + threadIdx.x) * 4;
    const float4 f = *(const float4*)(in + i);
    const v4bf p = {(bf16_t)f.x, (bf16_t)f.y, (bf16_t)f.z, (bf16_t)f.w};
    *(v4bf*)(out + i) = p;
}

// ---------------------------------------------------------------------------
// Embedding gather: x[s][d] = emb[ids[s]][d]
// ---------------------------------------------------------------------------
__global__ void embed_kernel(const int* __restrict__ ids,
                             const float* __restrict__ emb,
                             float* __restrict__ x) {
    int i = blockIdx.x * blockDim.x + threadIdx.x;
    int s = i / D_MODEL;
    int d = i - s * D_MODEL;
    x[i] = emb[(size_t)ids[s] * D_MODEL + d];
}

// ---------------------------------------------------------------------------
// RMSNorm over last dim (block per row); bf16 output feeds WMMA GEMM A.
// ---------------------------------------------------------------------------
template <int DW>
__global__ void rmsnorm_kernel(const float* __restrict__ x,
                               const float* __restrict__ w,
                               bf16_t* __restrict__ out) {
    __shared__ float red[256];
    const int s = blockIdx.x, tid = threadIdx.x;
    const float* row = x + (size_t)s * DW;
    float acc = 0.f;
    for (int d = tid; d < DW; d += 256) { float v = row[d]; acc += v * v; }
    red[tid] = acc;
    __syncthreads();
    for (int off = 128; off > 0; off >>= 1) {
        if (tid < off) red[tid] += red[tid + off];
        __syncthreads();
    }
    const float inv = rsqrtf(red[0] / (float)DW + 1e-5f);
    bf16_t* orow = out + (size_t)s * DW;
    for (int d = tid; d < DW; d += 256) orow[d] = (bf16_t)(row[d] * inv * w[d]);
}

// ---------------------------------------------------------------------------
// Tiled bf16 WMMA GEMM:  C[M,N] = epilogue( A[M,K] * W[N,K]^T )
//   bf16 operands in global; async-staged ping-pong LDS tiles; fp32 accum
//   via v_wmma_f32_16x16x32_bf16. Two barriers per K-slice.
// EPI: 0 = none, 1 = +bias then softplus, 2 = += resid
// ---------------------------------------------------------------------------
enum { EPI_NONE = 0, EPI_BIAS_SOFTPLUS = 1, EPI_RESID = 2 };

template <int BM, int BN, int WM, int WN, int EPI>
__global__ __launch_bounds__((BM / WM) * (BN / WN) * 32)
void gemm_bf16_wmma(const bf16_t* __restrict__ A,
                    const bf16_t* __restrict__ W,
                    float* __restrict__ C,
                    const float* __restrict__ bias,
                    const float* __restrict__ resid,
                    int M, int N, int Kd) {
    constexpr int BK   = 32;             // one WMMA K-slice per LDS tile
    constexpr int LDSS = BK + 8;         // padded bf16 row stride (80B)
    constexpr int NW   = (BM / WM) * (BN / WN);
    constexpr int NT   = NW * 32;
    constexpr int MI   = WM / 16;
    constexpr int NI   = WN / 16;
    constexpr int CA   = BM * BK / 8;    // 16B chunks per A tile
    constexpr int CB   = BN * BK / 8;

    __shared__ __align__(16) bf16_t sA[2][BM * LDSS];
    __shared__ __align__(16) bf16_t sB[2][BN * LDSS];

    const int tid  = threadIdx.x;
    const int lane = tid & 31;
    const int wave = tid >> 5;
    const int wrow = wave / (BN / WN);
    const int wcol = wave % (BN / WN);
    const int m0   = blockIdx.y * BM;
    const int n0   = blockIdx.x * BN;
    const int l15  = lane & 15;
    const int kh   = (lane & 16) ? 1 : 0;   // K-half select per ISA layout
    const int hi8  = (lane & 16) ? 8 : 0;   // C/D row offset for lanes 16..31

    v8f acc[MI][NI];
    const v8f vzero = {0.f, 0.f, 0.f, 0.f, 0.f, 0.f, 0.f, 0.f};
#pragma unroll
    for (int i = 0; i < MI; i++)
#pragma unroll
        for (int j = 0; j < NI; j++) acc[i][j] = vzero;

    // stage one K-slice (16-byte granules, per-lane global->LDS)
    auto stage = [&](int buf, int k0) {
#pragma unroll
        for (int c = tid; c < CA; c += NT) {
            const int r = c >> 2, q = c & 3;         // 4 chunks per 32-elem row
            stage16(A + (size_t)(m0 + r) * Kd + k0 + q * 8,
                    &sA[buf][r * LDSS + q * 8]);
        }
#pragma unroll
        for (int c = tid; c < CB; c += NT) {
            const int r = c >> 2, q = c & 3;
            stage16(W + (size_t)(n0 + r) * Kd + k0 + q * 8,
                    &sB[buf][r * LDSS + q * 8]);
        }
    };

    stage(0, 0);
    int buf = 0;
    for (int k0 = 0; k0 < Kd; k0 += BK, buf ^= 1) {
        stage_wait();
        __syncthreads();
        if (k0 + BK < Kd) stage(buf ^ 1, k0 + BK);

        // ---- fragments per ISA 7.12.2 16-bit layouts ----
        v16bf afrag[MI];
        v16bf bfrag[NI];
#pragma unroll
        for (int i = 0; i < MI; i++) {
            // A 16x32: lanes<16 -> K{0..7,16..23}; lanes>=16 -> K{8..15,24..31}
            const bf16_t* base = &sA[buf][(wrow * WM + i * 16 + l15) * LDSS];
            const v8bf lo = *(const v8bf*)(base + kh * 8);
            const v8bf hi = *(const v8bf*)(base + 16 + kh * 8);
            afrag[i] = __builtin_shufflevector(lo, hi, 0, 1, 2, 3, 4, 5, 6, 7,
                                               8, 9, 10, 11, 12, 13, 14, 15);
        }
#pragma unroll
        for (int j = 0; j < NI; j++) {
            // B 32x16: lane = column n; lanes<16 -> K{0..7,16..23}; lanes>=16 -> K{8..15,24..31}
            const bf16_t* base =
                &sB[buf][(wcol * WN + j * 16 + l15) * LDSS + kh * 8];
            const v8bf lo = *(const v8bf*)(base);
            const v8bf hi = *(const v8bf*)(base + 16);
            bfrag[j] = __builtin_shufflevector(lo, hi, 0, 1, 2, 3, 4, 5, 6, 7,
                                               8, 9, 10, 11, 12, 13, 14, 15);
        }
#pragma unroll
        for (int i = 0; i < MI; i++)
#pragma unroll
            for (int j = 0; j < NI; j++)
                acc[i][j] = __builtin_amdgcn_wmma_f32_16x16x32_bf16(
                    false, afrag[i], false, bfrag[j], (short)0, acc[i][j],
                    false, false);
        __syncthreads();
    }

    // ---- epilogue: C/D layout = VGPR v -> row (v + hi8), col = l15 ----
#pragma unroll
    for (int i = 0; i < MI; i++) {
#pragma unroll
        for (int j = 0; j < NI; j++) {
            const int col = n0 + wcol * WN + j * 16 + l15;
            const float b = (EPI == EPI_BIAS_SOFTPLUS) ? bias[col] : 0.f;
#pragma unroll
            for (int v = 0; v < 8; v++) {
                const int row = m0 + wrow * WM + i * 16 + hi8 + v;
                const size_t off = (size_t)row * N + col;
                float val = acc[i][j][v];
                if (EPI == EPI_BIAS_SOFTPLUS) val = softplus_f(val + b);
                else if (EPI == EPI_RESID)    val += resid[off];
                C[off] = val;
            }
        }
    }
}

// ---------------------------------------------------------------------------
// Depthwise causal conv1d (K=4) + bias + SiLU.
//   input = xz[:, 0:DI] (row stride 2*DI); outputs fp32 (for scan) + bf16
// ---------------------------------------------------------------------------
__global__ void conv_silu_kernel(const float* __restrict__ xz,
                                 const float* __restrict__ cw,
                                 const float* __restrict__ cb,
                                 float* __restrict__ xi,
                                 bf16_t* __restrict__ xi_bf) {
    const int i = blockIdx.x * blockDim.x + threadIdx.x;
    const int s = i / DI_DIM;
    const int ch = i - s * DI_DIM;
    float acc = cb[ch];
#pragma unroll
    for (int j = 0; j < K_CONV; j++) {
        const int sp = s - (K_CONV - 1) + j;
        if (sp >= 0)
            acc += cw[ch * K_CONV + j] * xz[(size_t)sp * (2 * DI_DIM) + ch];
    }
    const float v = silu_f(acc);
    xi[i] = v;
    xi_bf[i] = (bf16_t)v;
}

// ---------------------------------------------------------------------------
// Selective-scan: lane per channel, 16-entry state in registers.
// ---------------------------------------------------------------------------
__global__ void scan_kernel(const float* __restrict__ xi,
                            const float* __restrict__ dtb,
                            const float* __restrict__ bc,
                            const float* __restrict__ A_log,
                            const float* __restrict__ Dp,
                            float* __restrict__ yb) {
    const int ch = blockIdx.x * blockDim.x + threadIdx.x;
    float Aval[N_STATE], h[N_STATE];
#pragma unroll
    for (int n = 0; n < N_STATE; n++) {
        Aval[n] = -__expf(A_log[ch * N_STATE + n]);
        h[n] = 0.f;
    }
    const float dp = Dp[ch];
    for (int s = 0; s < S_LEN; s++) {
        const float u  = xi[(size_t)s * DI_DIM + ch];
        const float dt = dtb[(size_t)s * DI_DIM + ch];
        const float* bcs = bc + s * (2 * N_STATE);   // [B(16) | C(16)]
        float y = 0.f;
#pragma unroll
        for (int n = 0; n < N_STATE; n++) {
            const float dA = __expf(dt * Aval[n]);
            h[n] = dA * h[n] + (dt * bcs[n]) * u;
            y += h[n] * bcs[N_STATE + n];
        }
        yb[(size_t)s * DI_DIM + ch] = y + dp * u;
    }
}

// ---------------------------------------------------------------------------
// Gate: out_bf = (y * silu(z)) as bf16, z = xz[:, DI:2*DI]
// ---------------------------------------------------------------------------
__global__ void gate_kernel(const float* __restrict__ yb,
                            const float* __restrict__ xz,
                            bf16_t* __restrict__ yb_bf) {
    const int i = blockIdx.x * blockDim.x + threadIdx.x;
    const int s = i / DI_DIM;
    const int e = i - s * DI_DIM;
    const float z = xz[(size_t)s * (2 * DI_DIM) + DI_DIM + e];
    yb_bf[i] = (bf16_t)(yb[i] * silu_f(z));
}

// ---------------------------------------------------------------------------
extern "C" void kernel_launch(void* const* d_in, const int* in_sizes, int n_in,
                              void* d_out, int out_size, void* d_ws, size_t ws_size,
                              hipStream_t stream) {
    const int*   ids     = (const int*)  d_in[0];
    const float* emb     = (const float*)d_in[1];
    const float* in_w    = (const float*)d_in[2];
    const float* conv_w  = (const float*)d_in[3];
    const float* conv_b  = (const float*)d_in[4];
    const float* xp_w    = (const float*)d_in[5];
    const float* dt_w    = (const float*)d_in[6];
    const float* dt_b    = (const float*)d_in[7];
    const float* A_log   = (const float*)d_in[8];
    const float* Dp      = (const float*)d_in[9];
    const float* out_w   = (const float*)d_in[10];
    const float* norm_w  = (const float*)d_in[11];
    const float* normf_w = (const float*)d_in[12];
    const float* lm_w    = (const float*)d_in[13];
    float* logits = (float*)d_out;

    // ---- workspace layout ----
    float* x    = (float*)d_ws;                                  // S*D
    float* xz   = x    + (size_t)S_LEN * D_MODEL;                // S*2DI
    float* xi   = xz   + (size_t)S_LEN * 2 * DI_DIM;             // S*DI
    float* bcb  = xi   + (size_t)S_LEN * DI_DIM;                 // S*2N
    float* dtb  = bcb  + (size_t)S_LEN * 2 * N_STATE;            // S*DI
    float* yb   = dtb  + (size_t)S_LEN * DI_DIM;                 // S*DI
    bf16_t* hn_bf = (bf16_t*)(yb + (size_t)S_LEN * DI_DIM);      // S*D
    bf16_t* xi_bf = hn_bf + (size_t)S_LEN * D_MODEL;             // S*DI
    bf16_t* yb_bf = xi_bf + (size_t)S_LEN * DI_DIM;              // S*DI
    bf16_t* w_in  = yb_bf + (size_t)S_LEN * DI_DIM;              // NL*2DI*D
    bf16_t* w_xp  = w_in  + (size_t)N_LAYERS * 2 * DI_DIM * D_MODEL;
    bf16_t* w_dt  = w_xp  + (size_t)N_LAYERS * 2 * N_STATE * DI_DIM;
    bf16_t* w_out = w_dt  + (size_t)N_LAYERS * DI_DIM * DI_DIM;
    bf16_t* w_lm  = w_out + (size_t)N_LAYERS * D_MODEL * DI_DIM; // V*D

    // ---- one-shot weight conversion fp32 -> bf16 (L2-resident reuse) ----
    {
        const size_t n_in_w = (size_t)N_LAYERS * 2 * DI_DIM * D_MODEL;
        const size_t n_xp_w = (size_t)N_LAYERS * 2 * N_STATE * DI_DIM;
        const size_t n_dt_w = (size_t)N_LAYERS * DI_DIM * DI_DIM;
        const size_t n_out_w = (size_t)N_LAYERS * D_MODEL * DI_DIM;
        const size_t n_lm_w = (size_t)VOCAB * D_MODEL;
        cvt_bf16_kernel<<<n_in_w / 1024, 256, 0, stream>>>(in_w, w_in);
        cvt_bf16_kernel<<<n_xp_w / 1024, 256, 0, stream>>>(xp_w, w_xp);
        cvt_bf16_kernel<<<n_dt_w / 1024, 256, 0, stream>>>(dt_w, w_dt);
        cvt_bf16_kernel<<<n_out_w / 1024, 256, 0, stream>>>(out_w, w_out);
        cvt_bf16_kernel<<<n_lm_w / 1024, 256, 0, stream>>>(lm_w, w_lm);
    }

    embed_kernel<<<(S_LEN * D_MODEL) / 256, 256, 0, stream>>>(ids, emb, x);

    for (int l = 0; l < N_LAYERS; l++) {
        rmsnorm_kernel<D_MODEL><<<S_LEN, 256, 0, stream>>>(
            x, norm_w + (size_t)l * D_MODEL, hn_bf);

        gemm_bf16_wmma<128, 128, 32, 64, EPI_NONE>
            <<<dim3((2 * DI_DIM) / 128, S_LEN / 128), 256, 0, stream>>>(
                hn_bf, w_in + (size_t)l * 2 * DI_DIM * D_MODEL, xz,
                nullptr, nullptr, S_LEN, 2 * DI_DIM, D_MODEL);

        conv_silu_kernel<<<(S_LEN * DI_DIM) / 256, 256, 0, stream>>>(
            xz, conv_w + (size_t)l * DI_DIM * K_CONV,
            conv_b + (size_t)l * DI_DIM, xi, xi_bf);

        gemm_bf16_wmma<128, 32, 32, 32, EPI_NONE>
            <<<dim3((2 * N_STATE) / 32, S_LEN / 128), 128, 0, stream>>>(
                xi_bf, w_xp + (size_t)l * 2 * N_STATE * DI_DIM, bcb,
                nullptr, nullptr, S_LEN, 2 * N_STATE, DI_DIM);

        gemm_bf16_wmma<128, 128, 32, 64, EPI_BIAS_SOFTPLUS>
            <<<dim3(DI_DIM / 128, S_LEN / 128), 256, 0, stream>>>(
                xi_bf, w_dt + (size_t)l * DI_DIM * DI_DIM, dtb,
                dt_b + (size_t)l * DI_DIM, nullptr, S_LEN, DI_DIM, DI_DIM);

        scan_kernel<<<DI_DIM / 256, 256, 0, stream>>>(
            xi, dtb, bcb, A_log + (size_t)l * DI_DIM * N_STATE,
            Dp + (size_t)l * DI_DIM, yb);

        gate_kernel<<<(S_LEN * DI_DIM) / 256, 256, 0, stream>>>(yb, xz, yb_bf);

        gemm_bf16_wmma<128, 128, 32, 64, EPI_RESID>
            <<<dim3(D_MODEL / 128, S_LEN / 128), 256, 0, stream>>>(
                yb_bf, w_out + (size_t)l * D_MODEL * DI_DIM, x,
                nullptr, x, S_LEN, D_MODEL, DI_DIM);
    }

    rmsnorm_kernel<D_MODEL><<<S_LEN, 256, 0, stream>>>(x, normf_w, hn_bf);

    gemm_bf16_wmma<128, 128, 32, 64, EPI_NONE>
        <<<dim3(VOCAB / 128, S_LEN / 128), 256, 0, stream>>>(
            hn_bf, w_lm, logits, nullptr, nullptr, S_LEN, VOCAB, D_MODEL);
}
